// FourierPeriodNet_72069551227165
// MI455X (gfx1250) — compile-verified, matrix operand order checked by reading, not run on hardware
//
#include <hip/hip_runtime.h>

typedef __attribute__((ext_vector_type(16))) _Float16 v16h;
typedef __attribute__((ext_vector_type(2)))  _Float16 v2h;
typedef __attribute__((ext_vector_type(8)))  float    v8f;

#define WAVES   8
#define BLK_M   128            // rows per M-tile (WAVES * 16)
#define GM      4              // M-tiles per block (B staged once, reused)
#define BLK_N   256            // full channel dim per block -> no sin/cos dup
#define NT      (BLK_N / 16)   // 16 N-tiles
#define KSTEPS  8              // K = 256 (sin|cos concat), 32 per WMMA
#define C_TOT   256
#define F_TOT   128

// out[m, c] = sum_k A[m,k] * Bm[k,c]
//   A[m,k]  = k<128 ? sin(2*pi*t[m]*freqs[k]) : cos(2*pi*t[m]*freqs[k-128])
//   Bm[k,c] = k<128 ? amp_sin[c,k]            : amp_cos[c,k-128]
__global__ __launch_bounds__(256, 1)
void fourier_wmma_kernel(const float* __restrict__ t,
                         const float* __restrict__ freqs,
                         const float* __restrict__ amp_sin,
                         const float* __restrict__ amp_cos,
                         float* __restrict__ out)
{
    // Full 256x256 B matrix as f16, pre-swizzled into per-lane WMMA
    // fragment order: addr = (((nt*8 + kk)*32 + lane)*16 + j) f16.  128 KB.
    __shared__ _Float16 sB[NT * KSTEPS * 32 * 16];
    __shared__ float    sF[F_TOT];

    const int tid  = threadIdx.x;
    const int lane = tid & 31;
    const int wave = tid >> 5;

    if (tid < F_TOT) sF[tid] = freqs[tid];

    // ---- stage amp_sin/amp_cos (f32 -> f16) into fragment-ordered LDS ----
    // pair index p encodes: nt(4b) | kk(3b) | lane(5b) | jpair(3b);
    // the (k0,k1) pair is contiguous & even-aligned in the amp row -> float2.
    #pragma unroll 4
    for (int i = 0; i < 128; ++i) {
        int p  = tid + i * 256;            // 0 .. 32767 (32768 f16 pairs)
        int jp = p & 7;
        int l  = (p >> 3) & 31;
        int kk = (p >> 8) & 7;
        int nt = (p >> 11) & 15;
        int n  = nt * 16 + (l & 15);
        int j0 = jp * 2;
        int kb = kk * 32 + ((l >> 4) << 3);            // +8 for hi half-wave
        int k0 = kb + (j0 & 7) + ((j0 >> 3) << 4);     // even; k1 = k0+1
        const float* amp = (k0 < F_TOT) ? amp_sin : amp_cos;
        int f0 = k0 & (F_TOT - 1);
        float2 v = *(const float2*)(amp + n * F_TOT + f0);   // global b64
        v2h pk;
        pk.x = (_Float16)v.x;
        pk.y = (_Float16)v.y;
        *(v2h*)(&sB[p * 2]) = pk;                            // lds b32
    }
    __syncthreads();

    const int khalf = (lane >> 4) << 3;   // lanes 16-31 hold K+8 per A layout

    #pragma unroll 1
    for (int mt = 0; mt < GM; ++mt) {
        const int mbase = (blockIdx.x * GM + mt) * BLK_M + wave * 16;

        // ---- A fragments in registers via hardware V_SIN/V_COS ----
        // v_sin_f32/v_cos_f32 compute sin/cos(2*pi*x): feed x = t*f directly.
        const float tv = t[mbase + (lane & 15)];
        v16h a[KSTEPS];
        #pragma unroll
        for (int kk = 0; kk < KSTEPS; ++kk) {
            #pragma unroll
            for (int j = 0; j < 16; ++j) {
                int   k = kk * 32 + khalf + (j & 7) + ((j >> 3) << 4);
                float x = tv * sF[k & (F_TOT - 1)];
                float s = (k < F_TOT) ? __builtin_amdgcn_sinf(x)
                                      : __builtin_amdgcn_cosf(x);
                a[kk][j] = (_Float16)s;
            }
        }

        // ---- GEMM: per 16x16 N-tile, 8 chained WMMAs over K=256 ----
        const int mrow = mbase + khalf;   // D rows are mrow + r, r = 0..7
        #pragma unroll 1
        for (int nt = 0; nt < NT; ++nt) {
            v8f c = {0.f, 0.f, 0.f, 0.f, 0.f, 0.f, 0.f, 0.f};
            const _Float16* bp = &sB[(nt * KSTEPS) * 32 * 16 + lane * 16];
            #pragma unroll
            for (int kk = 0; kk < KSTEPS; ++kk) {
                v16h b = *(const v16h*)(bp + kk * 32 * 16);   // 2x ds_load_b128
                c = __builtin_amdgcn_wmma_f32_16x16x32_f16(
                        /*neg_a=*/false, a[kk], /*neg_b=*/false, b,
                        /*c_mod=*/(short)0, c, /*reuse_a=*/false, /*reuse_b=*/false);
            }
            float* op = out + (size_t)mrow * C_TOT + nt * 16 + (lane & 15);
            #pragma unroll
            for (int r = 0; r < 8; ++r)
                op[(size_t)r * C_TOT] = c[r];
        }
    }
}

extern "C" void kernel_launch(void* const* d_in, const int* in_sizes, int n_in,
                              void* d_out, int out_size, void* d_ws, size_t ws_size,
                              hipStream_t stream) {
    const float* t       = (const float*)d_in[0];   // (B*L) = 262144
    const float* freqs   = (const float*)d_in[1];   // (128)
    const float* amp_sin = (const float*)d_in[2];   // (256,128)
    const float* amp_cos = (const float*)d_in[3];   // (256,128)
    float*       out     = (float*)d_out;           // (B*L, 256)

    const int M = in_sizes[0];                      // 262144 = 512 * GM*BLK_M
    dim3 grid(M / (GM * BLK_M));
    fourier_wmma_kernel<<<grid, dim3(256), 0, stream>>>(t, freqs, amp_sin,
                                                        amp_cos, out);
}
